// GTrXL_26809185861961
// MI455X (gfx1250) — matrix operand order repaired and checked by reading, not compiled, not run on hardware
//
#include <hip/hip_runtime.h>
#include <hip/hip_bf16.h>

typedef __bf16 bf16_t;
typedef __attribute__((ext_vector_type(16))) __bf16 v16bf;
typedef __attribute__((ext_vector_type(8)))  float  v8f;

constexpr int kL  = 2;
constexpr int kB  = 64;
constexpr int kQ  = 128;
constexpr int kM  = 1024;
constexpr int kD  = 256;
constexpr int kH  = 4;
constexpr int kDH = 64;
constexpr int kF  = 1024;
constexpr int kK  = kM + kQ;      // 1152
constexpr int kBQ = kB * kQ;      // 8192
constexpr float kScale = 0.125f;  // 1/sqrt(DH)
constexpr float kEps   = 1e-6f;

// ---------------------------------------------------------------- helpers
__device__ __forceinline__ v16bf load2x16(const bf16_t* p0, const bf16_t* p1) {
  union { uint4 u[2]; v16bf v; } r;
  r.u[0] = *(const uint4*)p0;
  r.u[1] = *(const uint4*)p1;
  return r.v;
}

__device__ __forceinline__ v8f wmma_bf16(v16bf a, v16bf b, v8f c) {
  // D = A(16x32) * B(32x16) + C, f32 accumulate
  return __builtin_amdgcn_wmma_f32_16x16x32_bf16(false, a, false, b,
                                                 (short)0, c, false, false);
}

// ---------------------------------------------------------------- weight packing
// Pack f32 weight W[Kd][N] (row-major) into WMMA-B-tile layout (bf16):
// tile (nt, kt) of shape 32(K) x 16(N); per lane l: col = l&15, K-half = l>>4,
// 16 bf16: t<8 -> K = kt*32 + (l>>4)*8 + t; t>=8 -> K = kt*32 + 16 + (l>>4)*8 + t-8.
__global__ void pack_w_kernel(const float* __restrict__ W, bf16_t* __restrict__ Wp,
                              int Kd, int N) {
  int idx = blockIdx.x * 256 + threadIdx.x;
  int total = Kd * N;
  if (idx >= total) return;
  int tile = idx >> 9;        // 512 elements per tile
  int r    = idx & 511;
  int lane = r >> 4;
  int t    = r & 15;
  int kTiles = Kd >> 5;
  int nt = tile / kTiles;
  int kt = tile - nt * kTiles;
  int k = kt * 32 + (t & 8) * 2 + (lane >> 4) * 8 + (t & 7);
  int n = nt * 16 + (lane & 15);
  Wp[idx] = (bf16_t)W[(size_t)k * N + n];
}

// ---------------------------------------------------------------- GEMM
// C[rowsA, N] = A[rowsA, Kd](bf16 row-major) * W(packed bf16)
// 128 threads = 4 waves; block = 128 rows x 64 cols; wave w -> rows
// [blockM + w*32, +32). Weight panel staged in LDS via async-to-LDS DMA,
// double-buffered, shared by all 4 waves. 8 WMMAs per K-step per wave.
template <typename OutT>
__global__ __launch_bounds__(128) void gemm_bf16_kernel(
    const bf16_t* __restrict__ A, const bf16_t* __restrict__ Wp,
    OutT* __restrict__ C, int rowsA, int Kd, int N) {
  __shared__ bf16_t bpan[2][4][512];     // [buf][n-tile][tile data] = 2 x 4KB

  const int tid  = threadIdx.x;
  const int lane = tid & 31;
  const int wave = tid >> 5;
  const int lh = lane & 15, hh = lane >> 4;
  const int rowBase = blockIdx.y * 128 + wave * 32;
  const int ntBase  = blockIdx.x * 4;
  const int kTiles  = Kd >> 5;

  // wave w fills n-tile w of the panel: 2 async b128 per wave per panel
  const bf16_t* wsrc = Wp + ((size_t)(ntBase + wave) * kTiles) * 512 + lane * 16;
  auto issuePanel = [&](int kt, int buf) {
    unsigned long long src =
        (unsigned long long)(uintptr_t)(wsrc + (size_t)kt * 512);
    unsigned dst = (unsigned)(uintptr_t)&bpan[buf][wave][lane * 16];
    asm volatile("global_load_async_to_lds_b128 %0, %1, off"
                 :: "v"(dst), "v"(src) : "memory");
    asm volatile("global_load_async_to_lds_b128 %0, %1, off offset:16"
                 :: "v"(dst), "v"(src) : "memory");
  };

  v8f acc[2][4];
#pragma unroll
  for (int i = 0; i < 2; ++i)
#pragma unroll
    for (int j = 0; j < 4; ++j) acc[i][j] = (v8f){};

  const bf16_t* arow0 = A + (size_t)(rowBase + lh) * Kd + hh * 8;
  const bf16_t* arow1 = A + (size_t)(rowBase + 16 + lh) * Kd + hh * 8;

  issuePanel(0, 0);
  for (int kt = 0; kt < kTiles; ++kt) {
    const int buf = kt & 1;
    if (kt + 1 < kTiles) {
      issuePanel(kt + 1, buf ^ 1);
      asm volatile("s_wait_asynccnt 2" ::: "memory");  // panel kt landed (ours)
    } else {
      asm volatile("s_wait_asynccnt 0" ::: "memory");
    }
    __syncthreads();                     // every wave's tile of panel kt landed

    const bf16_t* ap0 = arow0 + kt * 32;
    const bf16_t* ap1 = arow1 + kt * 32;
    __builtin_prefetch(ap0 + 64, 0, 1);  // global_prefetch_b8
    v16bf a0 = load2x16(ap0, ap0 + 16);  // A-frag rows rowBase+lh
    v16bf a1 = load2x16(ap1, ap1 + 16);  // A-frag rows rowBase+16+lh
#pragma unroll
    for (int nt = 0; nt < 4; ++nt) {
      const bf16_t* bp = &bpan[buf][nt][lane * 16];
      v16bf b = load2x16(bp, bp + 8);    // ds_load_b128 x2
      acc[0][nt] = wmma_bf16(a0, b, acc[0][nt]);
      acc[1][nt] = wmma_bf16(a1, b, acc[1][nt]);
    }
    __syncthreads();                     // all waves done with buf before refill
  }

  const int colBase = ntBase * 16 + lh;
#pragma unroll
  for (int rg = 0; rg < 2; ++rg) {
#pragma unroll
    for (int g = 0; g < 8; ++g) {
      size_t r = (size_t)rowBase + rg * 16 + g + hh * 8;
      OutT* cp = C + r * N + colBase;
      cp[0]  = (OutT)acc[rg][0][g];
      cp[16] = (OutT)acc[rg][1][g];
      cp[32] = (OutT)acc[rg][2][g];
      cp[48] = (OutT)acc[rg][3][g];
    }
  }
}

// ---------------------------------------------------------------- fused attention
// One wave per (b, h, 16-row q block). Flash softmax over K=1152 keys.
// bd (rel-pos) term precomputed into LDS; rel-shift == shifted LDS read.
__global__ __launch_bounds__(32) void attn_kernel(
    const bf16_t* __restrict__ qw, const bf16_t* __restrict__ qr,
    const bf16_t* __restrict__ kmat, const bf16_t* __restrict__ vT,
    const bf16_t* __restrict__ rel, bf16_t* __restrict__ o) {
  __shared__ bf16_t bdlds[16 * kK];   // 36 KB
  __shared__ bf16_t pt[16 * 32];      // 1 KB, P tile C-layout -> A-layout relay

  const int qt = blockIdx.x, h = blockIdx.y, b = blockIdx.z;
  const int q0 = qt * 16;
  const int lane = threadIdx.x;
  const int lh = lane & 15, hh = lane >> 4;

  // Q fragments (A-layout: row = lane&15, K contiguous at hh*16 .. +15)
  const bf16_t* qwrow = qw + ((size_t)b * kQ + q0 + lh) * kD + h * kDH;
  const bf16_t* qrrow = qr + ((size_t)b * kQ + q0 + lh) * kD + h * kDH;
  v16bf qw0 = load2x16(qwrow + hh * 16,      qwrow + hh * 16 + 8);
  v16bf qw1 = load2x16(qwrow + 32 + hh * 16, qwrow + 40 + hh * 16);
  v16bf qr0 = load2x16(qrrow + hh * 16,      qrrow + hh * 16 + 8);
  v16bf qr1 = load2x16(qrrow + 32 + hh * 16, qrrow + 40 + hh * 16);

  // Phase 1: bd_pre[16][kK] = (q + r_r_bias) . rel^T  (per head)
  for (int t0 = 0; t0 < kK; t0 += 16) {
    const bf16_t* rp = rel + (size_t)(t0 + lh) * kD + h * kDH;
    v16bf rb0 = load2x16(rp + hh * 16,      rp + hh * 16 + 8);   // d 0..31
    v16bf rb1 = load2x16(rp + 32 + hh * 16, rp + 40 + hh * 16);  // d 32..63
    v8f acc = {};
    acc = wmma_bf16(qr0, rb0, acc);
    acc = wmma_bf16(qr1, rb1, acc);
    for (int g = 0; g < 8; ++g)
      bdlds[(size_t)(g + hh * 8) * kK + t0 + lh] = (bf16_t)acc[g];
  }
  __syncthreads();

  // Phase 2: flash attention over K in chunks of 32
  float mrun[8], lrun[8];
  v8f O0 = {}, O1 = {}, O2 = {}, O3 = {};
  for (int g = 0; g < 8; ++g) { mrun[g] = -1e30f; lrun[g] = 0.0f; }

  for (int kc = 0; kc < kK; kc += 32) {
    float s[2][8];
    for (int st = 0; st < 2; ++st) {
      const int j0 = kc + st * 16;
      const bf16_t* kp = kmat + ((size_t)b * kK + j0 + lh) * kD + h * kDH;
      v16bf kb0 = load2x16(kp + hh * 16,      kp + hh * 16 + 8);
      v16bf kb1 = load2x16(kp + 32 + hh * 16, kp + 40 + hh * 16);
      v8f acc = {};
      acc = wmma_bf16(qw0, kb0, acc);
      acc = wmma_bf16(qw1, kb1, acc);
      const int j = j0 + lh;
      for (int g = 0; g < 8; ++g) {
        const int i = q0 + g + hh * 8;
        int t = j - i + (kQ - 1);                 // rel-shift index, >=0 always
        t = (t < kK - 1) ? t : (kK - 1);          // clamp (masked anyway if OOB)
        float bd = (float)bdlds[(size_t)(g + hh * 8) * kK + t];
        float val = (acc[g] + bd) * kScale;
        s[st][g] = (j <= kM + i) ? val : -1e30f;  // causal mask
      }
    }
    // online softmax (row = g + hh*8; reduce across the 16 lanes of a half)
    float corr[8];
    for (int g = 0; g < 8; ++g) {
      float m = fmaxf(s[0][g], s[1][g]);
      for (int off = 1; off < 16; off <<= 1) m = fmaxf(m, __shfl_xor(m, off, 32));
      float mn = fmaxf(mrun[g], m);
      corr[g] = __expf(mrun[g] - mn);
      mrun[g] = mn;
      float p0 = __expf(s[0][g] - mn);
      float p1 = __expf(s[1][g] - mn);
      s[0][g] = p0; s[1][g] = p1;
      float ps = p0 + p1;
      for (int off = 1; off < 16; off <<= 1) ps += __shfl_xor(ps, off, 32);
      lrun[g] = lrun[g] * corr[g] + ps;
    }
    for (int g = 0; g < 8; ++g) {
      O0[g] *= corr[g]; O1[g] *= corr[g]; O2[g] *= corr[g]; O3[g] *= corr[g];
    }
    // P: C-layout -> LDS -> A-layout fragment
    __syncthreads();
    for (int st = 0; st < 2; ++st)
      for (int g = 0; g < 8; ++g)
        pt[(g + hh * 8) * 32 + st * 16 + lh] = (bf16_t)s[st][g];
    __syncthreads();
    const bf16_t* pr = pt + lh * 32 + hh * 16;
    v16bf pa = load2x16(pr, pr + 8);
    // V B-fragments from transposed V: vT[b][h][dh][k], contiguous in k
    const size_t vbh = ((size_t)b * kH + h) * kDH;
    const bf16_t* v0 = vT + (vbh + 0 * 16 + lh) * kK + kc + hh * 16;
    const bf16_t* v1 = vT + (vbh + 1 * 16 + lh) * kK + kc + hh * 16;
    const bf16_t* v2 = vT + (vbh + 2 * 16 + lh) * kK + kc + hh * 16;
    const bf16_t* v3 = vT + (vbh + 3 * 16 + lh) * kK + kc + hh * 16;
    O0 = wmma_bf16(pa, load2x16(v0, v0 + 8), O0);
    O1 = wmma_bf16(pa, load2x16(v1, v1 + 8), O1);
    O2 = wmma_bf16(pa, load2x16(v2, v2 + 8), O2);
    O3 = wmma_bf16(pa, load2x16(v3, v3 + 8), O3);
  }
  // finalize
  for (int g = 0; g < 8; ++g) {
    float inv = 1.0f / lrun[g];
    int row = q0 + g + hh * 8;
    bf16_t* op = o + ((size_t)b * kQ + row) * kD + h * kDH;
    op[0 * 16 + lh] = (bf16_t)(O0[g] * inv);
    op[1 * 16 + lh] = (bf16_t)(O1[g] * inv);
    op[2 * 16 + lh] = (bf16_t)(O2[g] * inv);
    op[3 * 16 + lh] = (bf16_t)(O3[g] * inv);
  }
}

// ---------------------------------------------------------------- VALU kernels
__global__ __launch_bounds__(32) void rmsnorm_kernel(const float* __restrict__ x,
                                                     const float* __restrict__ scale,
                                                     bf16_t* __restrict__ out) {
  const size_t row = blockIdx.x;
  const int lane = threadIdx.x;
  const float* xp = x + row * kD;
  float v[8]; float ss = 0.0f;
  for (int i = 0; i < 8; ++i) { v[i] = xp[lane + i * 32]; ss += v[i] * v[i]; }
  for (int off = 1; off < 32; off <<= 1) ss += __shfl_xor(ss, off, 32);
  float r = rsqrtf(ss * (1.0f / kD) + kEps);
  for (int i = 0; i < 8; ++i)
    out[row * kD + lane + i * 32] = (bf16_t)(v[i] * r * scale[lane + i * 32]);
}

__global__ void init_h_kernel(const float* __restrict__ x, float* __restrict__ h,
                              bf16_t* __restrict__ hb, int n) {
  int i = blockIdx.x * 256 + threadIdx.x;
  if (i >= n) return;
  float v = x[i];
  h[i] = v; hb[i] = (bf16_t)v;
}

__global__ void build_kv_kernel(const float* __restrict__ mems_l,
                                const bf16_t* __restrict__ hn,
                                bf16_t* __restrict__ kv, int n) {
  int i = blockIdx.x * 256 + threadIdx.x;
  if (i >= n) return;
  int d = i % kD;
  size_t bk = (size_t)i / kD;
  int k = (int)(bk % kK);
  int b = (int)(bk / kK);
  if (k < kM) kv[i] = (bf16_t)mems_l[((size_t)b * kM + k) * kD + d];
  else        kv[i] = hn[((size_t)b * kQ + (k - kM)) * kD + d];
}

__global__ void relbase_kernel(bf16_t* __restrict__ out, int n) {
  int i = blockIdx.x * 256 + threadIdx.x;
  if (i >= n) return;
  int d = i % kD;
  int p = i / kD;
  int f = d & 127;
  float inv = __expf(-(2.0f * f / (float)kD) * 9.210340371976184f); // 10000^{-2f/D}
  float sv = (float)(kK - 1 - p) * inv;
  out[i] = (bf16_t)((d < 128) ? __sinf(sv) : __cosf(sv));
}

__global__ void make_qwqr_kernel(const float* __restrict__ q,
                                 const float* __restrict__ rwb,
                                 const float* __restrict__ rrb,
                                 bf16_t* __restrict__ qw, bf16_t* __restrict__ qr,
                                 int n) {
  int i = blockIdx.x * 256 + threadIdx.x;
  if (i >= n) return;
  int d = i % kD;                 // d = h*DH + dd matches flat [H*DH] bias
  float v = q[i];
  qw[i] = (bf16_t)(v + rwb[d]);
  qr[i] = (bf16_t)(v + rrb[d]);
}

__global__ void transpose_v_kernel(const bf16_t* __restrict__ v,
                                   bf16_t* __restrict__ vT, int n) {
  int i = blockIdx.x * 256 + threadIdx.x;
  if (i >= n) return;
  int d = i % kD;
  size_t bk = (size_t)i / kD;
  int k = (int)(bk % kK);
  int b = (int)(bk / kK);
  int h = d / kDH, dd = d % kDH;
  vT[(((size_t)b * kH + h) * kDH + dd) * kK + k] = v[i];
}

__global__ void biascvt_kernel(float* __restrict__ y, const float* __restrict__ bias,
                               bf16_t* __restrict__ yb, int n, int dmod) {
  int i = blockIdx.x * 256 + threadIdx.x;
  if (i >= n) return;
  float v = y[i];
  if (bias) v += bias[i % dmod];
  y[i] = v;
  yb[i] = (bf16_t)v;
}

__global__ void gelu_kernel(const float* __restrict__ z, const float* __restrict__ bias,
                            bf16_t* __restrict__ out, int n, int fmod) {
  int i = blockIdx.x * 256 + threadIdx.x;
  if (i >= n) return;
  float v = z[i] + bias[i % fmod];
  float g = 0.5f * v * (1.0f + tanhf(0.7978845608028654f * (v + 0.044715f * v * v * v)));
  out[i] = (bf16_t)g;
}

__global__ void gate_kernel(const float* __restrict__ gx, const float* __restrict__ gy,
                            const float* __restrict__ bx, const float* __restrict__ y,
                            float* __restrict__ h, bf16_t* __restrict__ hb,
                            int n, int dmod) {
  int i = blockIdx.x * 256 + threadIdx.x;
  if (i >= n) return;
  float z = gx[i] + bx[i % dmod] + gy[i];
  float g = 1.0f / (1.0f + __expf(-z));
  float hv = g * y[i] + (1.0f - g) * h[i];
  h[i] = hv;
  hb[i] = (bf16_t)hv;
}

__global__ void copy_out_kernel(const float* __restrict__ h, float* __restrict__ out,
                                int n) {
  int i = blockIdx.x * 256 + threadIdx.x;
  if (i >= n) return;
  out[i] = h[i];
}

// ---------------------------------------------------------------- host
static inline int grd(int n) { return (n + 255) / 256; }

extern "C" void kernel_launch(void* const* d_in, const int* in_sizes, int n_in,
                              void* d_out, int out_size, void* d_ws, size_t ws_size,
                              hipStream_t stream) {
  const float* x       = (const float*)d_in[0];
  const float* mems    = (const float*)d_in[1];
  const float* wq      = (const float*)d_in[2];
  const float* wk      = (const float*)d_in[3];
  const float* wv      = (const float*)d_in[4];
  const float* wr      = (const float*)d_in[5];
  const float* wo      = (const float*)d_in[6];
  const float* rwb     = (const float*)d_in[7];
  const float* rrb     = (const float*)d_in[8];
  const float* attn_ns = (const float*)d_in[9];
  const float* ff_ns   = (const float*)d_in[10];
  const float* ffw1    = (const float*)d_in[11];
  const float* ffb1    = (const float*)d_in[12];
  const float* ffw2    = (const float*)d_in[13];
  const float* ffb2    = (const float*)d_in[14];
  const float* agwx    = (const float*)d_in[15];
  const float* agbx    = (const float*)d_in[16];
  const float* agwy    = (const float*)d_in[17];
  const float* fgwx    = (const float*)d_in[18];
  const float* fgbx    = (const float*)d_in[19];
  const float* fgwy    = (const float*)d_in[20];

  // workspace carve-out
  char* wsp = (char*)d_ws;
  auto alloc = [&](size_t bytes) -> void* {
    void* p = (void*)wsp;
    wsp += (bytes + 255) & ~(size_t)255;
    return p;
  };
  float*  h      = (float*)alloc((size_t)kBQ * kD * 4);
  bf16_t* h_bf   = (bf16_t*)alloc((size_t)kBQ * kD * 2);
  bf16_t* hn_bf  = (bf16_t*)alloc((size_t)kBQ * kD * 2);
  bf16_t* kv_bf  = (bf16_t*)alloc((size_t)kB * kK * kD * 2);
  bf16_t* k_bf   = (bf16_t*)alloc((size_t)kB * kK * kD * 2);
  bf16_t* v_bf   = (bf16_t*)alloc((size_t)kB * kK * kD * 2);
  bf16_t* vT_bf  = (bf16_t*)alloc((size_t)kB * kK * kD * 2);
  float*  q_f    = (float*)alloc((size_t)kBQ * kD * 4);
  bf16_t* qw_bf  = (bf16_t*)alloc((size_t)kBQ * kD * 2);
  bf16_t* qr_bf  = (bf16_t*)alloc((size_t)kBQ * kD * 2);
  bf16_t* rb_bf  = (bf16_t*)alloc((size_t)kK * kD * 2);
  bf16_t* rel_bf = (bf16_t*)alloc((size_t)kK * kD * 2);
  bf16_t* o_bf   = (bf16_t*)alloc((size_t)kBQ * kD * 2);
  float*  y_f    = (float*)alloc((size_t)kBQ * kD * 4);
  bf16_t* y_bf   = (bf16_t*)alloc((size_t)kBQ * kD * 2);
  float*  gx_f   = (float*)alloc((size_t)kBQ * kD * 4);
  float*  gy_f   = (float*)alloc((size_t)kBQ * kD * 4);
  float*  ff1_f  = (float*)alloc((size_t)kBQ * kF * 4);
  bf16_t* ff1_bf = (bf16_t*)alloc((size_t)kBQ * kF * 2);
  bf16_t* wq_p   = (bf16_t*)alloc((size_t)kL * kD * kD * 2);
  bf16_t* wk_p   = (bf16_t*)alloc((size_t)kL * kD * kD * 2);
  bf16_t* wv_p   = (bf16_t*)alloc((size_t)kL * kD * kD * 2);
  bf16_t* wr_p   = (bf16_t*)alloc((size_t)kL * kD * kD * 2);
  bf16_t* wo_p   = (bf16_t*)alloc((size_t)kL * kD * kD * 2);
  bf16_t* agx_p  = (bf16_t*)alloc((size_t)kL * kD * kD * 2);
  bf16_t* agy_p  = (bf16_t*)alloc((size_t)kL * kD * kD * 2);
  bf16_t* fgx_p  = (bf16_t*)alloc((size_t)kL * kD * kD * 2);
  bf16_t* fgy_p  = (bf16_t*)alloc((size_t)kL * kD * kD * 2);
  bf16_t* w1_p   = (bf16_t*)alloc((size_t)kL * kD * kF * 2);
  bf16_t* w2_p   = (bf16_t*)alloc((size_t)kL * kF * kD * 2);

  auto packW = [&](const float* W, bf16_t* Wp, int Kd, int N) {
    int n = Kd * N;
    pack_w_kernel<<<grd(n), 256, 0, stream>>>(W, Wp, Kd, N);
  };
  for (int l = 0; l < kL; ++l) {
    size_t dd = (size_t)l * kD * kD;
    packW(wq + dd,   wq_p + dd,  kD, kD);
    packW(wk + dd,   wk_p + dd,  kD, kD);
    packW(wv + dd,   wv_p + dd,  kD, kD);
    packW(wr + dd,   wr_p + dd,  kD, kD);
    packW(wo + dd,   wo_p + dd,  kD, kD);
    packW(agwx + dd, agx_p + dd, kD, kD);
    packW(agwy + dd, agy_p + dd, kD, kD);
    packW(fgwx + dd, fgx_p + dd, kD, kD);
    packW(fgwy + dd, fgy_p + dd, kD, kD);
    packW(ffw1 + (size_t)l * kD * kF, w1_p + (size_t)l * kD * kF, kD, kF);
    packW(ffw2 + (size_t)l * kF * kD, w2_p + (size_t)l * kF * kD, kF, kD);
  }
  relbase_kernel<<<grd(kK * kD), 256, 0, stream>>>(rb_bf, kK * kD);
  init_h_kernel<<<grd(kBQ * kD), 256, 0, stream>>>(x, h, h_bf, kBQ * kD);

  const int nQD  = kBQ * kD;
  const int nKVD = kB * kK * kD;
  const dim3 gemmBlk(128);
  for (int l = 0; l < kL; ++l) {
    const size_t dd = (size_t)l * kD * kD;
    // ---- attention sub-block
    rmsnorm_kernel<<<kBQ, 32, 0, stream>>>(h, attn_ns + l * kD, hn_bf);
    build_kv_kernel<<<grd(nKVD), 256, 0, stream>>>(
        mems + (size_t)l * kB * kM * kD, hn_bf, kv_bf, nKVD);
    gemm_bf16_kernel<float><<<dim3(kD / 64, kBQ / 128), gemmBlk, 0, stream>>>(
        hn_bf, wq_p + dd, q_f, kBQ, kD, kD);
    gemm_bf16_kernel<bf16_t><<<dim3(kD / 64, (kB * kK) / 128), gemmBlk, 0, stream>>>(
        kv_bf, wk_p + dd, k_bf, kB * kK, kD, kD);
    gemm_bf16_kernel<bf16_t><<<dim3(kD / 64, (kB * kK) / 128), gemmBlk, 0, stream>>>(
        kv_bf, wv_p + dd, v_bf, kB * kK, kD, kD);
    transpose_v_kernel<<<grd(nKVD), 256, 0, stream>>>(v_bf, vT_bf, nKVD);
    gemm_bf16_kernel<bf16_t><<<dim3(kD / 64, kK / 128), gemmBlk, 0, stream>>>(
        rb_bf, wr_p + dd, rel_bf, kK, kD, kD);
    make_qwqr_kernel<<<grd(nQD), 256, 0, stream>>>(
        q_f, rwb + (size_t)l * kH * kDH, rrb + (size_t)l * kH * kDH,
        qw_bf, qr_bf, nQD);
    attn_kernel<<<dim3(kQ / 16, kH, kB), 32, 0, stream>>>(
        qw_bf, qr_bf, k_bf, vT_bf, rel_bf, o_bf);
    gemm_bf16_kernel<float><<<dim3(kD / 64, kBQ / 128), gemmBlk, 0, stream>>>(
        o_bf, wo_p + dd, y_f, kBQ, kD, kD);
    biascvt_kernel<<<grd(nQD), 256, 0, stream>>>(y_f, nullptr, y_bf, nQD, kD);
    gemm_bf16_kernel<float><<<dim3(kD / 64, kBQ / 128), gemmBlk, 0, stream>>>(
        h_bf, agx_p + dd, gx_f, kBQ, kD, kD);
    gemm_bf16_kernel<float><<<dim3(kD / 64, kBQ / 128), gemmBlk, 0, stream>>>(
        y_bf, agy_p + dd, gy_f, kBQ, kD, kD);
    gate_kernel<<<grd(nQD), 256, 0, stream>>>(
        gx_f, gy_f, agbx + l * kD, y_f, h, h_bf, nQD, kD);
    // ---- feed-forward sub-block
    rmsnorm_kernel<<<kBQ, 32, 0, stream>>>(h, ff_ns + l * kD, hn_bf);
    gemm_bf16_kernel<float><<<dim3(kF / 64, kBQ / 128), gemmBlk, 0, stream>>>(
        hn_bf, w1_p + (size_t)l * kD * kF, ff1_f, kBQ, kD, kF);
    gelu_kernel<<<grd(kBQ * kF), 256, 0, stream>>>(
        ff1_f, ffb1 + l * kF, ff1_bf, kBQ * kF, kF);
    gemm_bf16_kernel<float><<<dim3(kD / 64, kBQ / 128), gemmBlk, 0, stream>>>(
        ff1_bf, w2_p + (size_t)l * kF * kD, y_f, kBQ, kF, kD);
    biascvt_kernel<<<grd(nQD), 256, 0, stream>>>(y_f, ffb2 + l * kD, y_bf, nQD, kD);
    gemm_bf16_kernel<float><<<dim3(kD / 64, kBQ / 128), gemmBlk, 0, stream>>>(
        h_bf, fgx_p + dd, gx_f, kBQ, kD, kD);
    gemm_bf16_kernel<float><<<dim3(kD / 64, kBQ / 128), gemmBlk, 0, stream>>>(
        y_bf, fgy_p + dd, gy_f, kBQ, kD, kD);
    gate_kernel<<<grd(nQD), 256, 0, stream>>>(
        gx_f, gy_f, fgbx + l * kD, y_f, h, h_bf, nQD, kD);
  }
  copy_out_kernel<<<grd(nQD), 256, 0, stream>>>(h, (float*)d_out, nQD);
  (void)in_sizes; (void)n_in; (void)out_size; (void)ws_size;
}
